// SimpleStructureModule_5918464933924
// MI455X (gfx1250) — matrix operand order repaired and checked by reading, not compiled, not run on hardware
//
#include <hip/hip_runtime.h>

// ---------------- constants ----------------
#define BB 4
#define LL 768
#define DD 256
#define HH 8
#define PP 64
#define FFD 1024
#define DK 32
#define MROWS (BB*LL)          // 3072

typedef __attribute__((ext_vector_type(16))) __bf16 v16bf;
typedef __attribute__((ext_vector_type(8)))  float  v8f;

union FragU { v16bf v; uint4 u[2]; };

__device__ inline unsigned short f2bf(float f) {
  union { float f; unsigned u; } x; x.f = f;
  unsigned u = x.u;
  return (unsigned short)((u + 0x7fffu + ((u >> 16) & 1u)) >> 16);
}

// A-fragment (16-bit, 16x32): per lane two 8-elem chunks at K = half*8 and half*8+16
__device__ inline v16bf load_afrag(const unsigned short* rowK0, int half) {
  FragU f;
  const unsigned short* p = rowK0 + half * 8;
  f.u[0] = *reinterpret_cast<const uint4*>(p);
  f.u[1] = *reinterpret_cast<const uint4*>(p + 16);
  return f.v;
}

// B-fragment (16-bit, 32x16 as B^T rows): per lane two 8-elem chunks at K = half*8 and half*8+16
__device__ inline v16bf load_bfrag(const unsigned short* rowK0, int half) {
  FragU f;
  const unsigned short* p = rowK0 + half * 8;
  f.u[0] = *reinterpret_cast<const uint4*>(p);
  f.u[1] = *reinterpret_cast<const uint4*>(p + 16);
  return f.v;
}

__device__ inline v8f wmma_bf16(v16bf a, v16bf b, v8f c) {
  return __builtin_amdgcn_wmma_f32_16x16x32_bf16(false, a, false, b, (short)0, c, false, false);
}

__device__ inline float wredsum32(float v) {
  for (int m = 16; m >= 1; m >>= 1) v += __shfl_xor(v, m, 32);
  return v;
}
__device__ inline float hredmax16(float v) {
  for (int m = 8; m >= 1; m >>= 1) v = fmaxf(v, __shfl_xor(v, m, 32));
  return v;
}
__device__ inline float hredsum16(float v) {
  for (int m = 8; m >= 1; m >>= 1) v += __shfl_xor(v, m, 32);
  return v;
}

// ---------------- weight transpose + bf16 convert: in fp32 [K][N] -> out bf16 [N][K] ----------------
__global__ __launch_bounds__(256) void k_transpose_bf16(const float* __restrict__ in,
                                                        unsigned short* __restrict__ out,
                                                        int K, int N) {
  int t = blockIdx.x * 256 + threadIdx.x;
  if (t >= K * N) return;
  int n = t / K, k = t % K;
  out[t] = f2bf(in[k * N + n]);
}

__global__ __launch_bounds__(256) void k_f2bf(const float* __restrict__ in,
                                              unsigned short* __restrict__ out, int n) {
  int t = blockIdx.x * 256 + threadIdx.x;
  if (t < n) out[t] = f2bf(in[t]);
}

// w_pair (64x8 fp32) -> transposed padded bf16 [16][64] (heads 8..15 = 0)
__global__ __launch_bounds__(256) void k_prep_wpT(const float* __restrict__ wp,
                                                  unsigned short* __restrict__ wpT) {
  int t = blockIdx.x * 256 + threadIdx.x;   // 0..1023
  int n = t / PP, k = t % PP;
  wpT[t] = (n < HH) ? f2bf(wp[k * HH + n]) : (unsigned short)0;
}

// dm[i][j] = sqrt(clip(|pred_t[i]-pred_t[j]|^2, 2, 37^2)); 2.36 MB, lives in L2,
// reused across all B*H bias planes
__global__ __launch_bounds__(256) void k_dm(const float* __restrict__ predt,
                                            float* __restrict__ dmLL) {
  int j = blockIdx.x * 256 + threadIdx.x;   // grid (3, 768)
  int i = blockIdx.y;
  float dx = predt[i * 3 + 0] - predt[j * 3 + 0];
  float dy = predt[i * 3 + 1] - predt[j * 3 + 1];
  float dz = predt[i * 3 + 2] - predt[j * 3 + 2];
  float sq = fminf(fmaxf(dx * dx + dy * dy + dz * dz, 2.0f), 1369.0f);
  dmLL[i * LL + j] = __builtin_amdgcn_sqrtf(sq);
}

// ---------------- pair bias (WMMA): wave handles 16 pairs; LN over P=64 with one
// partner-lane exchange; projection to heads via 2x v_wmma; distance bias fused ----------------
// grid: x = j-tiles (12 blocks of 4 waves), y = b*L + i
__global__ __launch_bounds__(128) void k_pairbias(const float* __restrict__ pf,
                                                  const float* __restrict__ dmLL,
                                                  const unsigned short* __restrict__ wpT,
                                                  const float* __restrict__ png,
                                                  const float* __restrict__ pnb,
                                                  const float* __restrict__ wdist,
                                                  float* __restrict__ bias) {
  int tid = threadIdx.x, wv = tid >> 5, l = tid & 31;
  int half = l >> 4, ln = l & 15;
  int j0 = (blockIdx.x * 4 + wv) * 16;
  int ybi = blockIdx.y;                 // b*LL + i
  int i = ybi % LL;
  int b = ybi / LL;

  // lane (ln, half) holds row (j0+ln), features at half*8 + c*16 + {0..7}, c=0..3
  const float* rowp = pf + ((long long)ybi * LL + j0 + ln) * PP + half * 8;
  __builtin_prefetch(rowp + 64 * PP, 0, 3);   // own row, 64 pairs (4 groups) ahead
  float x[32], pg[32], pb[32];
  for (int c = 0; c < 4; c++) {
    float4 lo = *reinterpret_cast<const float4*>(rowp + c * 16);
    float4 hi = *reinterpret_cast<const float4*>(rowp + c * 16 + 4);
    x[c*8+0]=lo.x; x[c*8+1]=lo.y; x[c*8+2]=lo.z; x[c*8+3]=lo.w;
    x[c*8+4]=hi.x; x[c*8+5]=hi.y; x[c*8+6]=hi.z; x[c*8+7]=hi.w;
    const float* gp = png + half * 8 + c * 16;
    const float* bp = pnb + half * 8 + c * 16;
    float4 g0 = *reinterpret_cast<const float4*>(gp);
    float4 g1 = *reinterpret_cast<const float4*>(gp + 4);
    float4 b0 = *reinterpret_cast<const float4*>(bp);
    float4 b1 = *reinterpret_cast<const float4*>(bp + 4);
    pg[c*8+0]=g0.x; pg[c*8+1]=g0.y; pg[c*8+2]=g0.z; pg[c*8+3]=g0.w;
    pg[c*8+4]=g1.x; pg[c*8+5]=g1.y; pg[c*8+6]=g1.z; pg[c*8+7]=g1.w;
    pb[c*8+0]=b0.x; pb[c*8+1]=b0.y; pb[c*8+2]=b0.z; pb[c*8+3]=b0.w;
    pb[c*8+4]=b1.x; pb[c*8+5]=b1.y; pb[c*8+6]=b1.z; pb[c*8+7]=b1.w;
  }
  // LN over 64: this lane has 32 features, partner (xor 16) has the other 32
  float s = 0.f;
  for (int i2 = 0; i2 < 32; i2++) s += x[i2];
  float mean = (s + __shfl_xor(s, 16, 32)) * (1.0f / 64.0f);
  float q = 0.f;
  for (int i2 = 0; i2 < 32; i2++) { float d = x[i2] - mean; q += d * d; }
  float var = (q + __shfl_xor(q, 16, 32)) * (1.0f / 64.0f);
  float rs = rsqrtf(var + 1e-5f);

  // normalize, scale, convert to bf16 A-fragments (a0: K 0..31, a1: K 32..63)
  v16bf a0, a1;
  for (int i2 = 0; i2 < 16; i2++) {
    a0[i2] = (__bf16)((x[i2]      - mean) * rs * pg[i2]      + pb[i2]);
    a1[i2] = (__bf16)((x[16 + i2] - mean) * rs * pg[16 + i2] + pb[16 + i2]);
  }
  v16bf b0 = load_bfrag(wpT + ln * PP, half);
  v16bf b1 = load_bfrag(wpT + ln * PP + 32, half);
  v8f c = {0.f,0.f,0.f,0.f,0.f,0.f,0.f,0.f};
  c = wmma_bf16(a0, b0, c);
  c = wmma_bf16(a1, b1, c);

  // distance bias from precomputed dm: 8 consecutive rows, broadcast float4 loads
  const float* dmr = dmLL + i * LL + j0 + half * 8;
  float4 d0 = *reinterpret_cast<const float4*>(dmr);
  float4 d1 = *reinterpret_cast<const float4*>(dmr + 4);
  float wd = (ln < HH) ? wdist[ln] : 0.f;
  c[0] += d0.x * wd; c[1] += d0.y * wd; c[2] += d0.z * wd; c[3] += d0.w * wd;
  c[4] += d1.x * wd; c[5] += d1.y * wd; c[6] += d1.z * wd; c[7] += d1.w * wd;

  // store: col ln = head, rows r = consecutive j -> two float4 per (head) lane
  if (ln < HH) {
    float* o = bias + ((b * HH + ln) * LL + i) * LL + j0 + half * 8;
    *reinterpret_cast<float4*>(o)     = make_float4(c[0], c[1], c[2], c[3]);
    *reinterpret_cast<float4*>(o + 4) = make_float4(c[4], c[5], c[6], c[7]);
  }
}

// ---------------- QKV GEMM (wave per 16x32 tile); V written transposed [B,H,32,L] ----------------
__global__ __launch_bounds__(32) void k_qkv(const unsigned short* __restrict__ Abf,
                                            const unsigned short* __restrict__ wqT,
                                            const unsigned short* __restrict__ wkT,
                                            const unsigned short* __restrict__ wvT,
                                            unsigned short* __restrict__ Qo,
                                            unsigned short* __restrict__ Ko,
                                            unsigned short* __restrict__ VT) {
  __shared__ __align__(16) unsigned short vt[16][32];
  int l = threadIdx.x, half = l >> 4, ln = l & 15;
  int m0 = blockIdx.x * 16, c0 = blockIdx.y * 32, which = blockIdx.z;
  const unsigned short* W = (which == 0) ? wqT : (which == 1) ? wkT : wvT;
  const unsigned short* arow = Abf + (m0 + ln) * DD;
  const unsigned short* w0 = W + (c0 + ln) * DD;
  const unsigned short* w1 = W + (c0 + 16 + ln) * DD;
  v8f acc0 = {0.f,0.f,0.f,0.f,0.f,0.f,0.f,0.f};
  v8f acc1 = {0.f,0.f,0.f,0.f,0.f,0.f,0.f,0.f};
  for (int kb = 0; kb < DD; kb += 32) {
    v16bf a  = load_afrag(arow + kb, half);
    v16bf b0 = load_bfrag(w0 + kb, half);
    v16bf b1 = load_bfrag(w1 + kb, half);
    acc0 = wmma_bf16(a, b0, acc0);
    acc1 = wmma_bf16(a, b1, acc1);
  }
  int bidx = m0 / LL, l0 = m0 % LL, h = c0 >> 5;
  if (which < 2) {
    unsigned short* O = (which == 0) ? Qo : Ko;
    for (int r = 0; r < 8; r++) {
      int lrow = l0 + r + half * 8;
      unsigned short* o = O + ((bidx * HH + h) * LL + lrow) * DK;
      o[ln]      = f2bf(acc0[r]);
      o[16 + ln] = f2bf(acc1[r]);
    }
  } else {
    for (int r = 0; r < 8; r++) {
      vt[r + half * 8][ln]      = f2bf(acc0[r]);
      vt[r + half * 8][16 + ln] = f2bf(acc1[r]);
    }
    __syncthreads();
    int dk = l;  // 0..31
    unsigned short* o = VT + ((bidx * HH + h) * DK + dk) * LL + l0;
    unsigned wbuf[8];
    for (int rr = 0; rr < 8; rr++)
      wbuf[rr] = (unsigned)vt[2 * rr][dk] | ((unsigned)vt[2 * rr + 1][dk] << 16);
    *reinterpret_cast<uint4*>(o)     = make_uint4(wbuf[0], wbuf[1], wbuf[2], wbuf[3]);
    *reinterpret_cast<uint4*>(o + 8) = make_uint4(wbuf[4], wbuf[5], wbuf[6], wbuf[7]);
  }
}

// ---------------- flash attention: wave per (b,h,16 queries), 32-key tiles ----------------
__global__ __launch_bounds__(32) void k_attn(const unsigned short* __restrict__ Q,
                                             const unsigned short* __restrict__ Kc,
                                             const unsigned short* __restrict__ VT,
                                             const float* __restrict__ bias,
                                             const int* __restrict__ mask,
                                             unsigned short* __restrict__ Obf) {
  __shared__ __align__(16) unsigned short P[16 * 32];
  int l = threadIdx.x, half = l >> 4, ln = l & 15;
  int b = blockIdx.z, h = blockIdx.y, q0 = blockIdx.x * 16;
  int bh = b * HH + h;
  v16bf aq = load_afrag(Q + (bh * LL + q0 + ln) * DK, half);
  float rowm[8], rowl[8];
  int qm[8];
  v8f acc0 = {0.f,0.f,0.f,0.f,0.f,0.f,0.f,0.f};
  v8f acc1 = {0.f,0.f,0.f,0.f,0.f,0.f,0.f,0.f};
  for (int r = 0; r < 8; r++) {
    rowm[r] = -3.0e38f; rowl[r] = 0.f;
    qm[r] = mask[b * LL + q0 + r + half * 8];
  }
  const float scale = 0.17677669529663687f;  // 1/sqrt(32)
  for (int j0 = 0; j0 < LL; j0 += 32) {
    v16bf bk0 = load_bfrag(Kc + (bh * LL + j0 + ln) * DK, half);
    v16bf bk1 = load_bfrag(Kc + (bh * LL + j0 + 16 + ln) * DK, half);
    v8f z = {0.f,0.f,0.f,0.f,0.f,0.f,0.f,0.f};
    v8f s0 = wmma_bf16(aq, bk0, z);
    v8f s1 = wmma_bf16(aq, bk1, z);
    int cm0 = mask[b * LL + j0 + ln];
    int cm1 = mask[b * LL + j0 + 16 + ln];
    for (int r = 0; r < 8; r++) {
      int row = q0 + r + half * 8;
      const float* bb = bias + (bh * LL + row) * LL + j0;
      float s0r = s0[r] * scale + bb[ln];
      float s1r = s1[r] * scale + bb[16 + ln];
      if (!(cm0 && qm[r])) s0r = -1e9f;
      if (!(cm1 && qm[r])) s1r = -1e9f;
      float tm = hredmax16(fmaxf(s0r, s1r));
      float mnew = fmaxf(rowm[r], tm);
      float corr = __expf(rowm[r] - mnew);
      float p0 = __expf(s0r - mnew);
      float p1 = __expf(s1r - mnew);
      float rsum = hredsum16(p0 + p1);
      rowl[r] = rowl[r] * corr + rsum;
      rowm[r] = mnew;
      acc0[r] *= corr;
      acc1[r] *= corr;
      P[(r + half * 8) * 32 + ln]      = f2bf(p0);
      P[(r + half * 8) * 32 + 16 + ln] = f2bf(p1);
    }
    __syncthreads();
    v16bf pa  = load_afrag(&P[ln * 32], half);
    v16bf pv0 = load_bfrag(VT + (bh * DK + ln) * LL + j0, half);
    v16bf pv1 = load_bfrag(VT + (bh * DK + 16 + ln) * LL + j0, half);
    acc0 = wmma_bf16(pa, pv0, acc0);
    acc1 = wmma_bf16(pa, pv1, acc1);
    __syncthreads();
  }
  for (int r = 0; r < 8; r++) {
    int row = q0 + r + half * 8;
    float inv = 1.0f / rowl[r];
    unsigned short* o = Obf + (b * LL + row) * DD + h * DK;
    o[ln]      = f2bf(acc0[r] * inv);
    o[16 + ln] = f2bf(acc1[r] * inv);
  }
}

// ---------------- generic bf16 GEMM, wave per 16x32 tile; epilogue modes ----------------
// mode 0: store fp32 raw; mode 1: +bias, exact GELU, store bf16; mode 2: +bias, store fp32
__global__ __launch_bounds__(32) void k_gemm_bf16(const unsigned short* __restrict__ A,
                                                  const unsigned short* __restrict__ Wt,
                                                  const float* __restrict__ bvec,
                                                  int M, int N, int K, int mode,
                                                  float* __restrict__ outf,
                                                  unsigned short* __restrict__ outb) {
  int l = threadIdx.x, half = l >> 4, ln = l & 15;
  int m0 = blockIdx.x * 16, c0 = blockIdx.y * 32;
  const unsigned short* arow = A + (m0 + ln) * K;
  const unsigned short* w0 = Wt + (c0 + ln) * K;
  const unsigned short* w1 = Wt + (c0 + 16 + ln) * K;
  v8f acc0 = {0.f,0.f,0.f,0.f,0.f,0.f,0.f,0.f};
  v8f acc1 = {0.f,0.f,0.f,0.f,0.f,0.f,0.f,0.f};
  for (int kb = 0; kb < K; kb += 32) {
    v16bf a  = load_afrag(arow + kb, half);
    v16bf b0 = load_bfrag(w0 + kb, half);
    v16bf b1 = load_bfrag(w1 + kb, half);
    acc0 = wmma_bf16(a, b0, acc0);
    acc1 = wmma_bf16(a, b1, acc1);
  }
  int cA = c0 + ln, cB = c0 + 16 + ln;
  for (int r = 0; r < 8; r++) {
    int m = m0 + r + half * 8;
    float vA = acc0[r], vB = acc1[r];
    if (mode == 0) {
      outf[m * N + cA] = vA;
      outf[m * N + cB] = vB;
    } else if (mode == 1) {
      float xa = vA + bvec[cA], xb = vB + bvec[cB];
      xa = 0.5f * xa * (1.0f + erff(xa * 0.70710678118654752f));
      xb = 0.5f * xb * (1.0f + erff(xb * 0.70710678118654752f));
      outb[m * N + cA] = f2bf(xa);
      outb[m * N + cB] = f2bf(xb);
    } else {
      outf[m * N + cA] = vA + bvec[cA];
      outf[m * N + cB] = vB + bvec[cB];
    }
  }
}

// ---------------- post-attention: src2 = LN(Z+src, ln_attn, 1e-6); srcA = LN(src+src2, n1, 1e-5) ----------------
__global__ __launch_bounds__(32) void k_post_attn_ln(const float* __restrict__ Z,
                                                     const float* __restrict__ src,
                                                     const float* __restrict__ lag,
                                                     const float* __restrict__ lab,
                                                     const float* __restrict__ n1g,
                                                     const float* __restrict__ n1b,
                                                     float* __restrict__ srcA,
                                                     unsigned short* __restrict__ srcAbf) {
  int row = blockIdx.x, l = threadIdx.x, c0 = l * 8;
  const float* z = Z + row * DD;
  const float* s = src + row * DD;
  float t[8], sv[8];
  for (int i = 0; i < 8; i++) { sv[i] = s[c0 + i]; t[i] = z[c0 + i] + sv[i]; }
  float sum = 0.f;
  for (int i = 0; i < 8; i++) sum += t[i];
  float m = wredsum32(sum) * (1.0f / DD);
  float q = 0.f;
  for (int i = 0; i < 8; i++) { float d = t[i] - m; q += d * d; }
  float rs = rsqrtf(wredsum32(q) * (1.0f / DD) + 1e-6f);
  float u[8];
  for (int i = 0; i < 8; i++) {
    float s2 = (t[i] - m) * rs * lag[c0 + i] + lab[c0 + i];
    u[i] = sv[i] + s2;
  }
  sum = 0.f;
  for (int i = 0; i < 8; i++) sum += u[i];
  float m2 = wredsum32(sum) * (1.0f / DD);
  q = 0.f;
  for (int i = 0; i < 8; i++) { float d = u[i] - m2; q += d * d; }
  float rs2 = rsqrtf(wredsum32(q) * (1.0f / DD) + 1e-5f);
  for (int i = 0; i < 8; i++) {
    float a = (u[i] - m2) * rs2 * n1g[c0 + i] + n1b[c0 + i];
    srcA[row * DD + c0 + i] = a;
    srcAbf[row * DD + c0 + i] = f2bf(a);
  }
}

// ---------------- final: srcB = LN(srcA+src2b, n2); out = LN(srcB+src2b, n3) ----------------
__global__ __launch_bounds__(32) void k_final_ln(const float* __restrict__ srcA,
                                                 const float* __restrict__ s2b,
                                                 const float* __restrict__ n2g,
                                                 const float* __restrict__ n2b,
                                                 const float* __restrict__ n3g,
                                                 const float* __restrict__ n3b,
                                                 float* __restrict__ out) {
  int row = blockIdx.x, l = threadIdx.x, c0 = l * 8;
  float v[8], r2[8];
  for (int i = 0; i < 8; i++) {
    r2[i] = s2b[row * DD + c0 + i];
    v[i] = srcA[row * DD + c0 + i] + r2[i];
  }
  float sum = 0.f;
  for (int i = 0; i < 8; i++) sum += v[i];
  float m = wredsum32(sum) * (1.0f / DD);
  float q = 0.f;
  for (int i = 0; i < 8; i++) { float d = v[i] - m; q += d * d; }
  float rs = rsqrtf(wredsum32(q) * (1.0f / DD) + 1e-5f);
  float w[8];
  for (int i = 0; i < 8; i++) {
    float sB = (v[i] - m) * rs * n2g[c0 + i] + n2b[c0 + i];
    w[i] = sB + r2[i];
  }
  sum = 0.f;
  for (int i = 0; i < 8; i++) sum += w[i];
  float m2 = wredsum32(sum) * (1.0f / DD);
  q = 0.f;
  for (int i = 0; i < 8; i++) { float d = w[i] - m2; q += d * d; }
  float rs2 = rsqrtf(wredsum32(q) * (1.0f / DD) + 1e-5f);
  for (int i = 0; i < 8; i++)
    out[row * DD + c0 + i] = (w[i] - m2) * rs2 * n3g[c0 + i] + n3b[c0 + i];
}

// ---------------- launcher ----------------
extern "C" void kernel_launch(void* const* d_in, const int* in_sizes, int n_in,
                              void* d_out, int out_size, void* d_ws, size_t ws_size,
                              hipStream_t stream) {
  const float* src   = (const float*)d_in[0];
  const float* pf    = (const float*)d_in[1];
  const float* predt = (const float*)d_in[2];
  const int*   mask  = (const int*)d_in[3];
  const float* wpair = (const float*)d_in[4];
  const float* png   = (const float*)d_in[5];
  const float* pnb   = (const float*)d_in[6];
  const float* wdist = (const float*)d_in[7];
  const float* wq    = (const float*)d_in[8];
  const float* wk    = (const float*)d_in[9];
  const float* wv    = (const float*)d_in[10];
  const float* wo    = (const float*)d_in[11];
  const float* lag   = (const float*)d_in[12];
  const float* lab   = (const float*)d_in[13];
  const float* w1    = (const float*)d_in[14];
  const float* b1    = (const float*)d_in[15];
  const float* w2    = (const float*)d_in[16];
  const float* b2    = (const float*)d_in[17];
  const float* n1g   = (const float*)d_in[18];
  const float* n1b   = (const float*)d_in[19];
  const float* n2g   = (const float*)d_in[20];
  const float* n2b   = (const float*)d_in[21];
  const float* n3g   = (const float*)d_in[22];
  const float* n3b   = (const float*)d_in[23];
  float* out = (float*)d_out;

  char* ws = (char*)d_ws;
  unsigned short* wqT    = (unsigned short*)(ws + 0);
  unsigned short* wkT    = (unsigned short*)(ws + 131072);
  unsigned short* wvT    = (unsigned short*)(ws + 262144);
  unsigned short* woT    = (unsigned short*)(ws + 393216);
  unsigned short* w1T    = (unsigned short*)(ws + 524288);
  unsigned short* w2T    = (unsigned short*)(ws + 1048576);
  unsigned short* srcbf  = (unsigned short*)(ws + 1572864);
  unsigned short* Qbf    = (unsigned short*)(ws + 3145728);
  unsigned short* Kbf    = (unsigned short*)(ws + 4718592);
  unsigned short* VTbf   = (unsigned short*)(ws + 6291456);
  unsigned short* attnbf = (unsigned short*)(ws + 7864320);
  float*          biasW  = (float*)(ws + 9437184);    // 75.5 MB
  float*          Z      = (float*)(ws + 84934656);
  float*          srcA   = (float*)(ws + 88080384);
  unsigned short* srcAbf = (unsigned short*)(ws + 91226112);
  unsigned short* midbf  = (unsigned short*)(ws + 92798976);
  float*          src2b  = (float*)(ws + 99090432);
  unsigned short* wpT    = (unsigned short*)(ws + 102236160);  // 2 KB
  float*          dmLL   = (float*)(ws + 102238208);           // 2.36 MB; end ~104.6 MB

  // 1. weight transposes + src convert + padded w_pair^T + dm table
  k_transpose_bf16<<<(DD*DD+255)/256, 256, 0, stream>>>(wq, wqT, DD, DD);
  k_transpose_bf16<<<(DD*DD+255)/256, 256, 0, stream>>>(wk, wkT, DD, DD);
  k_transpose_bf16<<<(DD*DD+255)/256, 256, 0, stream>>>(wv, wvT, DD, DD);
  k_transpose_bf16<<<(DD*DD+255)/256, 256, 0, stream>>>(wo, woT, DD, DD);
  k_transpose_bf16<<<(DD*FFD+255)/256, 256, 0, stream>>>(w1, w1T, DD, FFD);
  k_transpose_bf16<<<(FFD*DD+255)/256, 256, 0, stream>>>(w2, w2T, FFD, DD);
  k_f2bf<<<(MROWS*DD+255)/256, 256, 0, stream>>>(src, srcbf, MROWS*DD);
  k_prep_wpT<<<4, 256, 0, stream>>>(wpair, wpT);
  k_dm<<<dim3(LL/256, LL), 256, 0, stream>>>(predt, dmLL);

  // 2. pairwise + distance bias -> biasW (B,H,L,L); 16 pairs/wave, WMMA projection
  k_pairbias<<<dim3(LL/64, BB*LL), 128, 0, stream>>>(pf, dmLL, wpT, png, pnb, wdist, biasW);

  // 3. QKV projections (V transposed)
  k_qkv<<<dim3(MROWS/16, DD/32, 3), 32, 0, stream>>>(srcbf, wqT, wkT, wvT, Qbf, Kbf, VTbf);

  // 4. flash attention with pair bias + mask
  k_attn<<<dim3(LL/16, HH, BB), 32, 0, stream>>>(Qbf, Kbf, VTbf, biasW, mask, attnbf);

  // 5. out @ wo -> Z
  k_gemm_bf16<<<dim3(MROWS/16, DD/32), 32, 0, stream>>>(attnbf, woT, nullptr,
      MROWS, DD, DD, 0, Z, nullptr);

  // 6. src2 = LN(Z+src,1e-6); srcA = LN(src+src2,1e-5)
  k_post_attn_ln<<<MROWS, 32, 0, stream>>>(Z, src, lag, lab, n1g, n1b, srcA, srcAbf);

  // 7. mid = gelu(srcA @ w1 + b1)
  k_gemm_bf16<<<dim3(MROWS/16, FFD/32), 32, 0, stream>>>(srcAbf, w1T, b1,
      MROWS, FFD, DD, 1, nullptr, midbf);

  // 8. src2b = mid @ w2 + b2
  k_gemm_bf16<<<dim3(MROWS/16, DD/32), 32, 0, stream>>>(midbf, w2T, b2,
      MROWS, DD, FFD, 2, src2b, nullptr);

  // 9. srcB = LN(srcA+src2b, n2); out = LN(srcB+src2b, n3)
  k_final_ln<<<MROWS, 32, 0, stream>>>(srcA, src2b, n2g, n2b, n3g, n3b, out);
}
